// AttentionBlock3D_15255723836092
// MI455X (gfx1250) — compile-verified, matrix operand order checked by reading, not run on hardware
//
#include <hip/hip_runtime.h>

// ---------------------------------------------------------------------------
// AttentionBlock3D for MI455X (gfx1250, wave32, WMMA + Tensor Data Mover).
// GEMM-shaped math -> v_wmma_f32_16x16x32_f16 (f32 accum).
// Shared operand tiles (weight strips, K/V per head) staged into LDS via TDM.
// ---------------------------------------------------------------------------

typedef __attribute__((ext_vector_type(16))) _Float16 v16h;
typedef __attribute__((ext_vector_type(8)))  float    v8f;
typedef __attribute__((ext_vector_type(4)))  unsigned int u32x4;
typedef __attribute__((ext_vector_type(8)))  int          i32x8;
typedef __attribute__((ext_vector_type(4)))  int          i32x4;

#define WMMA(a, b, c) \
    __builtin_amdgcn_wmma_f32_16x16x32_f16(false, (a), false, (b), (short)0, (c), false, false)

#define HAVE_TDM __has_builtin(__builtin_amdgcn_tensor_load_to_lds)

constexpr int Cc   = 512;
constexpr int Nn   = 4096;          // 4*32*32
constexpr int NH   = 8;
constexpr int HD   = 64;
constexpr int O3   = 3 * Cc;        // 1536
constexpr int GRP  = 32;
constexpr int GSZ  = (Cc / GRP) * Nn;  // 65536 elems per group
constexpr int SC   = 128;           // flash attention s-chunk staged in LDS

#if HAVE_TDM
// Issue one TDM 2D tile load (D# groups per CDNA5 ISA ch.8): rows of
// `tile_w` f16 elements, `tile_h` rows, global row stride `stride` elements,
// packed contiguously into LDS at `ldst`. (6-arg builtin on this toolchain.)
__device__ __forceinline__ void tdm_load_2d(const _Float16* gsrc, _Float16* ldst,
                                            unsigned tile_w, unsigned tile_h,
                                            unsigned tensor_w, unsigned stride) {
    const unsigned lds_off = (unsigned)(unsigned long long)(size_t)ldst;
    const unsigned long long ga = (unsigned long long)(size_t)gsrc;
    u32x4 g0 = { 1u,                                   // count=1, user mode
                 lds_off,                              // lds_addr
                 (unsigned)(ga & 0xffffffffu),         // global_addr[31:0]
                 (unsigned)((ga >> 32) & 0x01ffffffu)  // global_addr[56:32]
                     | (2u << 30) };                   // type=2 ("image")
    i32x8 g1 = { (int)(1u << 16),                                    // data_size=2B
                 (int)((tensor_w & 0xffffu) << 16),                  // tensor_dim0 lo16
                 (int)(((tensor_w >> 16) & 0xffffu)
                       | ((tile_h & 0xffffu) << 16)),                // dim0 hi | tensor_dim1 lo
                 (int)((tile_w & 0xffffu) << 16),                    // tile_dim0
                 (int)(tile_h & 0xffffu),                            // tile_dim1
                 (int)stride,                                        // tensor_dim0_stride lo32
                 0, 0 };
    i32x4 g2 = {0, 0, 0, 0};
    i32x4 g3 = {0, 0, 0, 0};
    i32x8 g4 = {0, 0, 0, 0, 0, 0, 0, 0};
    __builtin_amdgcn_tensor_load_to_lds(g0, g1, g2, g3, g4, 0);
}
#else
// Fallback: cooperative vectorized copy (whole block participates).
__device__ __forceinline__ void coop_copy_2d(const _Float16* gsrc, _Float16* ldst,
                                             int tile_w, int tile_h, int stride) {
    const int cpr = tile_w / 8;                // 16B chunks per row
    const int total = cpr * tile_h;
    for (int i = threadIdx.x; i < total; i += 256) {
        const int r = i / cpr, ck = i % cpr;
        *(float4*)(ldst + r * tile_w + ck * 8) =
            *(const float4*)(gsrc + (size_t)r * stride + ck * 8);
    }
}
#endif

// ---------------------------------------------------------------------------
// 1) GroupNorm statistics: one block per group, LDS tree reduction.
// ---------------------------------------------------------------------------
__global__ void __launch_bounds__(256)
gn_stats_kernel(const float* __restrict__ x, float* __restrict__ stats) {
    __shared__ float r1[256];
    __shared__ float r2[256];
    const int g = blockIdx.x, tid = threadIdx.x;
    const float* base = x + (size_t)g * GSZ;
    float s = 0.f, s2 = 0.f;
    for (int i = tid; i < GSZ; i += 256) {
        float v = base[i];
        s += v; s2 += v * v;
    }
    r1[tid] = s; r2[tid] = s2;
    __syncthreads();
    for (int off = 128; off > 0; off >>= 1) {
        if (tid < off) { r1[tid] += r1[tid + off]; r2[tid] += r2[tid + off]; }
        __syncthreads();
    }
    if (tid == 0) {
        float mean = r1[0] / (float)GSZ;
        float var  = r2[0] / (float)GSZ - mean * mean;
        stats[2 * g]     = mean;
        stats[2 * g + 1] = rsqrtf(var + 1e-5f);
    }
}

// ---------------------------------------------------------------------------
// 2) Apply GN affine, write xn transposed [N][C] in f16 (B-fragment friendly).
// ---------------------------------------------------------------------------
__global__ void __launch_bounds__(256)
gn_apply_kernel(const float* __restrict__ x, const float* __restrict__ gw,
                const float* __restrict__ gb, const float* __restrict__ stats,
                _Float16* __restrict__ xnT) {
    const int idx = blockIdx.x * 256 + threadIdx.x;   // over C*N
    const int c = idx >> 12, n = idx & 4095, g = c >> 4;
    const float mean = stats[2 * g], rstd = stats[2 * g + 1];
    const float v = (x[idx] - mean) * rstd * gw[c] + gb[c];
    xnT[(size_t)n * Cc + c] = (_Float16)v;
}

// ---------------------------------------------------------------------------
// 3) Convert qkv_w (1536x512) and proj_w (512x512) to f16 (row-major kept).
// ---------------------------------------------------------------------------
__global__ void __launch_bounds__(256)
cvt_w_kernel(const float* __restrict__ qkvw, const float* __restrict__ projw,
             _Float16* __restrict__ qkvw16, _Float16* __restrict__ projw16) {
    const int idx = blockIdx.x * 256 + threadIdx.x;   // over O3*C + C*C
    if (idx < O3 * Cc) {
        qkvw16[idx] = (_Float16)qkvw[idx];
    } else {
        const int j = idx - O3 * Cc;
        projw16[j] = (_Float16)projw[j];
    }
}

// ---------------------------------------------------------------------------
// 4) QKV GEMM: D[1536x4096] = qkv_w[1536x512] * xn[512x4096].
//    Block: 8 waves share one 16-row A strip (16 KB, staged by TDM into LDS);
//    each wave computes a 16x64 strip. Epilogue routes into q^T (pre-scaled),
//    k^T [h][N][64] and v [h][64][N].
// ---------------------------------------------------------------------------
__global__ void __launch_bounds__(256)
qkv_gemm_kernel(const _Float16* __restrict__ A, const _Float16* __restrict__ B,
                const float* __restrict__ bias,
                _Float16* __restrict__ q16, _Float16* __restrict__ k16,
                _Float16* __restrict__ v16) {
    __shared__ __attribute__((aligned(64))) _Float16 ldsA[16 * Cc];   // 16 KB
    const int lane = threadIdx.x & 31, wave = threadIdx.x >> 5;
    const int lo = lane & 15, hi = lane >> 4;
    const int o0 = (blockIdx.x >> 3) * 16;                    // shared by block
    const int n0 = (((blockIdx.x & 7) << 3) + wave) * 64;     // per wave

#if HAVE_TDM
    if (threadIdx.x == 0)
        tdm_load_2d(A + (size_t)o0 * Cc, ldsA, 16 * Cc, 1, 16 * Cc, 16 * Cc);
    __builtin_amdgcn_s_wait_tensorcnt(0);
#else
    coop_copy_2d(A + (size_t)o0 * Cc, ldsA, 16 * Cc, 1, 16 * Cc);
#endif
    __syncthreads();

    v8f acc0 = {}, acc1 = {}, acc2 = {}, acc3 = {};
    const _Float16* arow = ldsA + lo * Cc + hi * 16;
    const _Float16* b0 = B + (size_t)(n0 +  0 + lo) * Cc + hi * 16;
    const _Float16* b1 = B + (size_t)(n0 + 16 + lo) * Cc + hi * 16;
    const _Float16* b2 = B + (size_t)(n0 + 32 + lo) * Cc + hi * 16;
    const _Float16* b3 = B + (size_t)(n0 + 48 + lo) * Cc + hi * 16;

    for (int kk = 0; kk < Cc; kk += 32) {
        const v16h a = *(const v16h*)(arow + kk);
        acc0 = WMMA(a, *(const v16h*)(b0 + kk), acc0);
        acc1 = WMMA(a, *(const v16h*)(b1 + kk), acc1);
        acc2 = WMMA(a, *(const v16h*)(b2 + kk), acc2);
        acc3 = WMMA(a, *(const v16h*)(b3 + kk), acc3);
    }

    const int kind = o0 >> 9;            // wave-uniform: q / k / v segment
    const int head = (o0 & 511) >> 6;
    const int dbase = o0 & 63;

    auto store_tile = [&](v8f acc, int t) {
#pragma unroll
        for (int j = 0; j < 8; ++j) {
            const int M = j + 8 * hi;
            const int n = n0 + t * 16 + lo;
            const int d = dbase + M;
            const float val = acc[j] + bias[o0 + M];
            if (kind == 0)       // q, fold softmax scale 1/sqrt(64)
                q16[((size_t)head * Nn + n) * HD + d] = (_Float16)(val * 0.125f);
            else if (kind == 1)  // k
                k16[((size_t)head * Nn + n) * HD + d] = (_Float16)val;
            else                 // v: [h][64][N]
                v16[((size_t)(head * HD + d)) * Nn + n] = (_Float16)val;
        }
    };
    store_tile(acc0, 0); store_tile(acc1, 1);
    store_tile(acc2, 2); store_tile(acc3, 3);
}

// ---------------------------------------------------------------------------
// 5) Flash attention: 8 waves/block own 16-row q tiles of ONE head and share
//    TDM-staged K/V s-chunks in LDS (8x L2 traffic reduction). Online softmax;
//    P bounced C-layout -> LDS -> A-layout (wave-synchronous DS round trip).
// ---------------------------------------------------------------------------
__global__ void __launch_bounds__(256)
flash_attn_kernel(const _Float16* __restrict__ qT, const _Float16* __restrict__ kT,
                  const _Float16* __restrict__ vM, _Float16* __restrict__ outT) {
    __shared__ __attribute__((aligned(64))) _Float16 ldsK[SC * HD];      // 16 KB [s][d]
    __shared__ __attribute__((aligned(64))) _Float16 ldsV[HD * SC];      // 16 KB [d][s]
    __shared__ __attribute__((aligned(32))) _Float16 plds[8][16 * 32];   // 8 KB
    const int lane = threadIdx.x & 31, wave = threadIdx.x >> 5;
    const int lo = lane & 15, hi = lane >> 4;
    const int head = blockIdx.x >> 5;                             // 32 blocks/head
    const int t0 = ((((blockIdx.x & 31) << 3) + wave) << 4);      // per-wave q tile

    const _Float16* qrow = qT + ((size_t)head * Nn + t0 + lo) * HD + hi * 16;
    const v16h qa0 = *(const v16h*)(qrow);        // K = d 0..31
    const v16h qa1 = *(const v16h*)(qrow + 32);   // K = d 32..63

    float m[8], lsum[8];
#pragma unroll
    for (int j = 0; j < 8; ++j) { m[j] = -3.0e38f; lsum[j] = 0.f; }
    v8f o0a = {}, o1a = {}, o2a = {}, o3a = {};
    _Float16* pl = &plds[wave][0];

    for (int sc = 0; sc < Nn; sc += SC) {
        __syncthreads();   // previous chunk fully consumed before restage (WAR)
#if HAVE_TDM
        if (threadIdx.x == 0) {
            // K rows sc..sc+SC of this head: contiguous 16 KB
            tdm_load_2d(kT + ((size_t)head * Nn + sc) * HD, ldsK,
                        SC * HD, 1, SC * HD, SC * HD);
            // V: 64 rows of SC elems, global row stride Nn
            tdm_load_2d(vM + (size_t)head * HD * Nn + sc, ldsV,
                        SC, HD, Nn, Nn);
        }
        __builtin_amdgcn_s_wait_tensorcnt(0);
#else
        coop_copy_2d(kT + ((size_t)head * Nn + sc) * HD, ldsK, SC * HD, 1, SC * HD);
        coop_copy_2d(vM + (size_t)head * HD * Nn + sc, ldsV, SC, HD, Nn);
#endif
        __syncthreads();

        for (int s0 = 0; s0 < SC; s0 += 32) {
            // --- S = (q*scale) @ k^T for 16 rows x 32 cols, K/V from LDS ---
            v8f scr[2];
#pragma unroll
            for (int half = 0; half < 2; ++half) {
                const _Float16* krow = ldsK + (s0 + half * 16 + lo) * HD + hi * 16;
                v8f c = {};
                c = WMMA(qa0, *(const v16h*)krow, c);
                c = WMMA(qa1, *(const v16h*)(krow + 32), c);
                scr[half] = c;
            }
            // --- online softmax (rows live in 16-lane halves) ---
            float corr[8];
#pragma unroll
            for (int j = 0; j < 8; ++j) {
                float mx = fmaxf(scr[0][j], scr[1][j]);
                mx = fmaxf(mx, __shfl_xor(mx, 1));
                mx = fmaxf(mx, __shfl_xor(mx, 2));
                mx = fmaxf(mx, __shfl_xor(mx, 4));
                mx = fmaxf(mx, __shfl_xor(mx, 8));
                const float mn = fmaxf(m[j], mx);
                corr[j] = __expf(m[j] - mn);
                m[j] = mn;
            }
#pragma unroll
            for (int j = 0; j < 8; ++j) {
                const float p0 = __expf(scr[0][j] - m[j]);
                const float p1 = __expf(scr[1][j] - m[j]);
                const int M = j + 8 * hi;
                pl[M * 32 + lo]      = (_Float16)p0;
                pl[M * 32 + 16 + lo] = (_Float16)p1;
                float rs = p0 + p1;
                rs += __shfl_xor(rs, 1);
                rs += __shfl_xor(rs, 2);
                rs += __shfl_xor(rs, 4);
                rs += __shfl_xor(rs, 8);
                lsum[j] = lsum[j] * corr[j] + rs;
                o0a[j] *= corr[j]; o1a[j] *= corr[j];
                o2a[j] *= corr[j]; o3a[j] *= corr[j];
            }
            // wave-synchronous LDS round trip (DS ops in-order per wave)
            __builtin_amdgcn_wave_barrier();
            asm volatile("s_wait_dscnt 0x0" ::: "memory");
            const v16h pa = *(const v16h*)(pl + lo * 32 + hi * 16);  // P 16x32 A-layout

            // --- O += P @ V_chunk (B cols = head dim, 4 tiles of 16) ---
            const _Float16* vbase = ldsV + s0 + hi * 16;
            o0a = WMMA(pa, *(const v16h*)(vbase + (lo)      * SC), o0a);
            o1a = WMMA(pa, *(const v16h*)(vbase + (16 + lo) * SC), o1a);
            o2a = WMMA(pa, *(const v16h*)(vbase + (32 + lo) * SC), o2a);
            o3a = WMMA(pa, *(const v16h*)(vbase + (48 + lo) * SC), o3a);
            __builtin_amdgcn_wave_barrier();
        }
    }

    // --- normalize and write out^T [N][C] f16 (B layout for proj GEMM) ---
#pragma unroll
    for (int j = 0; j < 8; ++j) {
        const float inv = 1.0f / lsum[j];
        const int M = j + 8 * hi;
        const size_t row = (size_t)(t0 + M) * Cc + (size_t)head * HD;
        outT[row +      lo] = (_Float16)(o0a[j] * inv);
        outT[row + 16 + lo] = (_Float16)(o1a[j] * inv);
        outT[row + 32 + lo] = (_Float16)(o2a[j] * inv);
        outT[row + 48 + lo] = (_Float16)(o3a[j] * inv);
    }
}

// ---------------------------------------------------------------------------
// 6) Proj GEMM + bias + residual, f32 output [C][N]. A strip staged via TDM.
// ---------------------------------------------------------------------------
__global__ void __launch_bounds__(256)
proj_gemm_kernel(const _Float16* __restrict__ A, const _Float16* __restrict__ B,
                 const float* __restrict__ bias, const float* __restrict__ x,
                 float* __restrict__ out) {
    __shared__ __attribute__((aligned(64))) _Float16 ldsA[16 * Cc];   // 16 KB
    const int lane = threadIdx.x & 31, wave = threadIdx.x >> 5;
    const int lo = lane & 15, hi = lane >> 4;
    const int o0 = (blockIdx.x >> 3) * 16;
    const int n0 = (((blockIdx.x & 7) << 3) + wave) * 64;

#if HAVE_TDM
    if (threadIdx.x == 0)
        tdm_load_2d(A + (size_t)o0 * Cc, ldsA, 16 * Cc, 1, 16 * Cc, 16 * Cc);
    __builtin_amdgcn_s_wait_tensorcnt(0);
#else
    coop_copy_2d(A + (size_t)o0 * Cc, ldsA, 16 * Cc, 1, 16 * Cc);
#endif
    __syncthreads();

    v8f acc0 = {}, acc1 = {}, acc2 = {}, acc3 = {};
    const _Float16* arow = ldsA + lo * Cc + hi * 16;
    const _Float16* b0 = B + (size_t)(n0 +  0 + lo) * Cc + hi * 16;
    const _Float16* b1 = B + (size_t)(n0 + 16 + lo) * Cc + hi * 16;
    const _Float16* b2 = B + (size_t)(n0 + 32 + lo) * Cc + hi * 16;
    const _Float16* b3 = B + (size_t)(n0 + 48 + lo) * Cc + hi * 16;

    for (int kk = 0; kk < Cc; kk += 32) {
        const v16h a = *(const v16h*)(arow + kk);
        acc0 = WMMA(a, *(const v16h*)(b0 + kk), acc0);
        acc1 = WMMA(a, *(const v16h*)(b1 + kk), acc1);
        acc2 = WMMA(a, *(const v16h*)(b2 + kk), acc2);
        acc3 = WMMA(a, *(const v16h*)(b3 + kk), acc3);
    }

    auto store_tile = [&](v8f acc, int t) {
#pragma unroll
        for (int j = 0; j < 8; ++j) {
            const int M = j + 8 * hi;
            const int n = n0 + t * 16 + lo;
            const size_t idx = (size_t)(o0 + M) * Nn + n;
            out[idx] = acc[j] + bias[o0 + M] + x[idx];
        }
    };
    store_tile(acc0, 0); store_tile(acc1, 1);
    store_tile(acc2, 2); store_tile(acc3, 3);
}

// ---------------------------------------------------------------------------
// Launch: workspace layout (all regions written every call, ~22 MB total)
// ---------------------------------------------------------------------------
extern "C" void kernel_launch(void* const* d_in, const int* in_sizes, int n_in,
                              void* d_out, int out_size, void* d_ws, size_t ws_size,
                              hipStream_t stream) {
    const float* x      = (const float*)d_in[0];
    const float* gn_w   = (const float*)d_in[1];
    const float* gn_b   = (const float*)d_in[2];
    const float* qkv_w  = (const float*)d_in[3];
    const float* qkv_b  = (const float*)d_in[4];
    const float* proj_w = (const float*)d_in[5];
    const float* proj_b = (const float*)d_in[6];
    float* out = (float*)d_out;

    char* ws = (char*)d_ws;
    size_t off = 0;
    float* stats = (float*)(ws + off);         off += 256;
    _Float16* xnT     = (_Float16*)(ws + off); off += (size_t)Nn * Cc * 2;   // 4 MB
    _Float16* qkvw16  = (_Float16*)(ws + off); off += (size_t)O3 * Cc * 2;   // 1.5 MB
    _Float16* projw16 = (_Float16*)(ws + off); off += (size_t)Cc * Cc * 2;   // 0.5 MB
    _Float16* q16     = (_Float16*)(ws + off); off += (size_t)Cc * Nn * 2;   // 4 MB
    _Float16* k16     = (_Float16*)(ws + off); off += (size_t)Cc * Nn * 2;   // 4 MB
    _Float16* v16     = (_Float16*)(ws + off); off += (size_t)Cc * Nn * 2;   // 4 MB
    _Float16* outT    = (_Float16*)(ws + off); off += (size_t)Nn * Cc * 2;   // 4 MB

    gn_stats_kernel<<<GRP, 256, 0, stream>>>(x, stats);
    gn_apply_kernel<<<(Cc * Nn) / 256, 256, 0, stream>>>(x, gn_w, gn_b, stats, xnT);
    cvt_w_kernel<<<(O3 * Cc + Cc * Cc) / 256, 256, 0, stream>>>(qkv_w, proj_w, qkvw16, projw16);
    qkv_gemm_kernel<<<(96 * 8), 256, 0, stream>>>(qkvw16, xnT, qkv_b, q16, k16, v16);
    flash_attn_kernel<<<(NH * 32), 256, 0, stream>>>(q16, k16, v16, outT);
    proj_gemm_kernel<<<(32 * 8), 256, 0, stream>>>(projw16, outT, proj_b, x, out);
}
